// GatedDeltaNetAttention_87316685128591
// MI455X (gfx1250) — matrix operand order repaired and checked by reading, not compiled, not run on hardware
//
#include <hip/hip_runtime.h>
#include <hip/hip_bf16.h>

// ---------------- problem constants ----------------
#define B_   4
#define T_   2048
#define C_   2048
#define H_   16
#define DK_  64
#define DV_  128
#define M_   (B_ * T_)          // 8192 rows
#define BHT_ (B_ * H_ * T_)     // 131072

typedef __attribute__((ext_vector_type(16))) __bf16 v16bf;
typedef __attribute__((ext_vector_type(8)))  float  v8f;

// round-to-nearest-even f32 -> bf16 (header-independent)
__device__ __forceinline__ __bf16 f2bf(float f) {
    unsigned int u;
    __builtin_memcpy(&u, &f, 4);
    unsigned int r = (u + 0x7FFFu + ((u >> 16) & 1u)) >> 16;
    unsigned short s = (unsigned short)r;
    __bf16 b;
    __builtin_memcpy(&b, &s, 2);
    return b;
}

// ---------------- fp32 -> bf16 conversion (float4 -> 4x bf16) ----------------
__global__ void cvt_f32_bf16(const float* __restrict__ in, __bf16* __restrict__ out, int n4) {
    int i = blockIdx.x * blockDim.x + threadIdx.x;
    if (i >= n4) return;
    float4 f = ((const float4*)in)[i];
    union { __bf16 b[4]; uint2 u; } o;
    o.b[0] = f2bf(f.x); o.b[1] = f2bf(f.y); o.b[2] = f2bf(f.z); o.b[3] = f2bf(f.w);
    ((uint2*)out)[i] = o.u;
}

// ---------------- WMMA bf16 GEMM:  C[M,N] = A[M,K] * W[N,K]^T ----------------
// ISA 7.12.2 fragment layouts (wave32):
//  A 16x32 bf16: lane l (m=l&15, h=l>>4) holds A[m][k0+8h .. +7] and A[m][k0+16+8h .. +7]
//  B 32x16 bf16: lane l (n=l&15, h=l>>4) holds W[n][k0+8h .. +7] and W[n][k0+16+8h .. +7]
//  C/D 16x16 f32: lane l element r -> C[r + 8*(l>>4)][l&15]
__device__ __forceinline__ v16bf load_a_frag(const __bf16* A, int ldk, int row, int k0, int lane) {
    int m = lane & 15, h = lane >> 4;
    const __bf16* p = A + (size_t)(row + m) * ldk + k0 + 8 * h;
    v16bf f;
    ((uint4*)&f)[0] = *(const uint4*)(p);        // K = k0+8h   .. +7
    ((uint4*)&f)[1] = *(const uint4*)(p + 16);   // K = k0+16+8h.. +7
    return f;
}
__device__ __forceinline__ v16bf load_b_frag(const __bf16* W, int ldk, int col, int k0, int lane) {
    int n = lane & 15, h = lane >> 4;
    const __bf16* p = W + (size_t)(col + n) * ldk + k0 + 8 * h;
    v16bf f;
    ((uint4*)&f)[0] = *(const uint4*)(p);        // K = k0+8h   .. +7
    ((uint4*)&f)[1] = *(const uint4*)(p + 16);   // K = k0+16+8h.. +7
    return f;
}

// 128 threads = 4 waves (2x2), each wave owns a 64x64 tile (4x4 WMMA tiles).
// Block tile 128x128. Operands are L2-resident (192MB L2 >> working set).
__global__ __launch_bounds__(128) void gemm_bf16_wmma(
        const __bf16* __restrict__ A, const __bf16* __restrict__ W,
        float* __restrict__ C, int M, int N, int K) {
    const int lane = threadIdx.x & 31;
    const int wave = threadIdx.x >> 5;
    const int bm = blockIdx.y * 128 + (wave >> 1) * 64;
    const int bn = blockIdx.x * 128 + (wave & 1) * 64;

    v8f acc[4][4];
    const v8f vzero = {0.f, 0.f, 0.f, 0.f, 0.f, 0.f, 0.f, 0.f};
#pragma unroll
    for (int i = 0; i < 4; ++i)
#pragma unroll
        for (int j = 0; j < 4; ++j) acc[i][j] = vzero;

    for (int k0 = 0; k0 < K; k0 += 32) {
        v16bf a[4], b[4];
#pragma unroll
        for (int i = 0; i < 4; ++i) a[i] = load_a_frag(A, K, bm + 16 * i, k0, lane);
#pragma unroll
        for (int j = 0; j < 4; ++j) b[j] = load_b_frag(W, K, bn + 16 * j, k0, lane);
#pragma unroll
        for (int i = 0; i < 4; ++i)
#pragma unroll
            for (int j = 0; j < 4; ++j)
                acc[i][j] = __builtin_amdgcn_wmma_f32_16x16x32_bf16(
                    false, a[i], false, b[j], (short)0, acc[i][j], false, false);
    }

    const int n0 = lane & 15;
    const int r0 = (lane >> 4) * 8;
#pragma unroll
    for (int i = 0; i < 4; ++i)
#pragma unroll
        for (int j = 0; j < 4; ++j)
#pragma unroll
            for (int r = 0; r < 8; ++r)
                C[(size_t)(bm + 16 * i + r0 + r) * N + (bn + 16 * j + n0)] = acc[i][j][r];
}

// ---------------- beta / g small projections (fp32) ----------------
__global__ void beta_g_kernel(const float* __restrict__ x,
                              const float* __restrict__ a_w, const float* __restrict__ b_w,
                              const float* __restrict__ b_bias, const float* __restrict__ beta_bias,
                              const float* __restrict__ A_log, const float* __restrict__ dt_bias,
                              float* __restrict__ beta, float* __restrict__ g) {
    int tid = blockIdx.x * blockDim.x + threadIdx.x;
    if (tid >= M_ * 32) return;
    int m = tid >> 5, idx = tid & 31, h = idx & 15, isg = idx >> 4;
    const float4* xr = (const float4*)(x + (size_t)m * C_);
    const float4* wr = (const float4*)((isg ? a_w : b_w) + (size_t)h * C_);
    float s0 = 0.f, s1 = 0.f, s2 = 0.f, s3 = 0.f;
    for (int i = 0; i < C_ / 4; ++i) {
        float4 a = xr[i], w = wr[i];
        s0 = fmaf(a.x, w.x, s0);
        s1 = fmaf(a.y, w.y, s1);
        s2 = fmaf(a.z, w.z, s2);
        s3 = fmaf(a.w, w.w, s3);
    }
    float s = (s0 + s1) + (s2 + s3);
    if (isg) {
        float z = s + dt_bias[h];
        float sp = (z > 20.f) ? z : log1pf(expf(z));
        g[(size_t)m * H_ + h] = -expf(A_log[h]) * sp;
    } else {
        float z = s + b_bias[h] + beta_bias[h];
        beta[(size_t)m * H_ + h] = 2.f / (1.f + expf(-z));
    }
}

// ---------------- causal depthwise conv (width 4) + SiLU ----------------
// in: (B,T,inC) fp32, channels [chOff, chOff+Cc). out: (B,H,T,Dd), Dd = Cc/H.
__global__ void dwconv_silu(const float* __restrict__ in, const float* __restrict__ cw,
                            float* __restrict__ out, int inC, int chOff, int Cc, int Dd) {
    long long gid = (long long)blockIdx.x * blockDim.x + threadIdx.x;
    if (gid >= (long long)B_ * T_ * Cc) return;
    int c = (int)(gid % Cc);
    int t = (int)((gid / Cc) % T_);
    int b = (int)(gid / ((long long)Cc * T_));
    float acc = 0.f;
#pragma unroll
    for (int j = 0; j < 4; ++j) {
        int ti = t - 3 + j;
        if (ti >= 0)
            acc += in[((size_t)(b * T_ + ti)) * inC + chOff + c] * cw[c * 4 + j];
    }
    float y = acc / (1.f + expf(-acc));   // silu
    int h = c / Dd, d = c % Dd;
    out[(((size_t)(b * H_ + h) * T_ + t)) * Dd + d] = y;
}

// ---------------- l2 norm over DK=64 (in place), optional SCALE ----------------
__global__ void l2norm_rows(float* __restrict__ data, int rows, float scale) {
    int r = blockIdx.x * blockDim.x + threadIdx.x;
    if (r >= rows) return;
    float4* p = (float4*)(data + (size_t)r * DK_);
    float4 v[16];
    float s = 0.f;
#pragma unroll
    for (int i = 0; i < 16; ++i) {
        v[i] = p[i];
        s += v[i].x * v[i].x + v[i].y * v[i].y + v[i].z * v[i].z + v[i].w * v[i].w;
    }
    float inv = rsqrtf(s + 1e-6f) * scale;
#pragma unroll
    for (int i = 0; i < 16; ++i) {
        v[i].x *= inv; v[i].y *= inv; v[i].z *= inv; v[i].w *= inv;
        p[i] = v[i];
    }
}

// ---------------- gated delta-rule recurrence ----------------
// One block per (b,h). 128 threads; thread v owns state column S[0..63][v]
// in 64 VGPRs. k_t/q_t double-buffered in LDS (one barrier/step, next step's
// operands staged during current compute). Reductions use 4 independent FMA
// chains to avoid a 64-deep serial dependency (latency -> issue bound).
__global__ __launch_bounds__(128) void delta_recurrence(
        const float* __restrict__ qc, const float* __restrict__ kc,
        const float* __restrict__ vc, const float* __restrict__ g,
        const float* __restrict__ beta, float* __restrict__ orec) {
    const int bh = blockIdx.x;          // 0..63
    const int b = bh >> 4, h = bh & 15;
    const int v = threadIdx.x;          // 0..127
    __shared__ float sk[2][DK_];
    __shared__ float sq[2][DK_];

    float S[DK_];
#pragma unroll
    for (int i = 0; i < DK_; ++i) S[i] = 0.f;

    const float* kp = kc + (size_t)bh * T_ * DK_;
    const float* qp = qc + (size_t)bh * T_ * DK_;
    const float* vp = vc + (size_t)bh * T_ * DV_;
    const float* gp = g + (size_t)b * T_ * H_ + h;
    const float* bp = beta + (size_t)b * T_ * H_ + h;
    float* op = orec + (size_t)bh * T_ * DV_;

    // stage t = 0
    if (threadIdx.x < DK_) sk[0][threadIdx.x] = kp[threadIdx.x];
    else                   sq[0][threadIdx.x - DK_] = qp[threadIdx.x - DK_];
    __syncthreads();

    for (int t = 0; t < T_; ++t) {
        const int cur = t & 1, nxt = cur ^ 1;
        // stage next step's k/q while computing this step
        if (t + 1 < T_) {
            if (threadIdx.x < DK_)
                sk[nxt][threadIdx.x] = kp[(size_t)(t + 1) * DK_ + threadIdx.x];
            else
                sq[nxt][threadIdx.x - DK_] = qp[(size_t)(t + 1) * DK_ + threadIdx.x - DK_];
        }
        float vt = vp[(size_t)t * DV_ + v];
        float eg = expf(gp[(size_t)t * H_]);
        float bt = bp[(size_t)t * H_];

        float kreg[DK_];
        float m0 = 0.f, m1 = 0.f, m2 = 0.f, m3 = 0.f;
#pragma unroll
        for (int i = 0; i < DK_; i += 4) {
            float k0 = sk[cur][i + 0], k1 = sk[cur][i + 1];
            float k2 = sk[cur][i + 2], k3 = sk[cur][i + 3];
            kreg[i + 0] = k0; kreg[i + 1] = k1; kreg[i + 2] = k2; kreg[i + 3] = k3;
            S[i + 0] *= eg; S[i + 1] *= eg; S[i + 2] *= eg; S[i + 3] *= eg;
            m0 = fmaf(k0, S[i + 0], m0);
            m1 = fmaf(k1, S[i + 1], m1);
            m2 = fmaf(k2, S[i + 2], m2);
            m3 = fmaf(k3, S[i + 3], m3);
        }
        float mem = (m0 + m1) + (m2 + m3);
        float delta = (vt - mem) * bt;

        float o0 = 0.f, o1 = 0.f, o2 = 0.f, o3 = 0.f;
#pragma unroll
        for (int i = 0; i < DK_; i += 4) {
            S[i + 0] = fmaf(kreg[i + 0], delta, S[i + 0]);
            S[i + 1] = fmaf(kreg[i + 1], delta, S[i + 1]);
            S[i + 2] = fmaf(kreg[i + 2], delta, S[i + 2]);
            S[i + 3] = fmaf(kreg[i + 3], delta, S[i + 3]);
            o0 = fmaf(sq[cur][i + 0], S[i + 0], o0);
            o1 = fmaf(sq[cur][i + 1], S[i + 1], o1);
            o2 = fmaf(sq[cur][i + 2], S[i + 2], o2);
            o3 = fmaf(sq[cur][i + 3], S[i + 3], o3);
        }
        op[(size_t)t * DV_ + v] = (o0 + o1) + (o2 + o3);
        __syncthreads();   // cur buffer fully consumed before it is re-staged
    }
}

// ---------------- RMS norm over DV + SiLU(g_out) gate, emit bf16 ----------------
__global__ void rms_gate(const float* __restrict__ orec, const float* __restrict__ vg,
                         __bf16* __restrict__ gated) {
    int r = blockIdx.x * blockDim.x + threadIdx.x;   // (b,h,t) row
    if (r >= BHT_) return;
    int t = r % T_, h = (r / T_) % H_, b = r / (T_ * H_);
    const float* orow = orec + (size_t)r * DV_;
    const float* grow = vg + ((size_t)(b * T_ + t)) * 4096 + 2048 + (size_t)h * DV_;
    __bf16* out = gated + ((size_t)(b * T_ + t)) * (H_ * DV_) + (size_t)h * DV_;

    float s0 = 0.f, s1 = 0.f, s2 = 0.f, s3 = 0.f;
    for (int i = 0; i < DV_; i += 4) {
        float x0 = orow[i], x1 = orow[i + 1], x2 = orow[i + 2], x3 = orow[i + 3];
        s0 = fmaf(x0, x0, s0); s1 = fmaf(x1, x1, s1);
        s2 = fmaf(x2, x2, s2); s3 = fmaf(x3, x3, s3);
    }
    float inv = rsqrtf(((s0 + s1) + (s2 + s3)) * (1.f / DV_) + 1.1920929e-07f);
    for (int i = 0; i < DV_; ++i) {
        float gv = grow[i];
        float gate = gv / (1.f + expf(-gv));
        out[i] = f2bf(orow[i] * inv * gate);
    }
}

// ---------------- host launcher ----------------
extern "C" void kernel_launch(void* const* d_in, const int* in_sizes, int n_in,
                              void* d_out, int out_size, void* d_ws, size_t ws_size,
                              hipStream_t stream) {
    (void)in_sizes; (void)n_in; (void)out_size; (void)ws_size;
    const float* x        = (const float*)d_in[0];
    const float* qk_w     = (const float*)d_in[1];
    const float* vg_w     = (const float*)d_in[2];
    const float* o_w      = (const float*)d_in[3];
    const float* a_w      = (const float*)d_in[4];
    const float* b_w      = (const float*)d_in[5];
    const float* b_bias   = (const float*)d_in[6];
    const float* beta_bias= (const float*)d_in[7];
    const float* A_log    = (const float*)d_in[8];
    const float* dt_bias  = (const float*)d_in[9];
    const float* q_conv_w = (const float*)d_in[10];
    const float* k_conv_w = (const float*)d_in[11];
    const float* v_conv_w = (const float*)d_in[12];
    float* out = (float*)d_out;

    // workspace carve-up (256B aligned)
    size_t off = 0;
    char* base = (char*)d_ws;
    auto nxt = [&](size_t bytes) -> char* {
        char* p = base + off;
        off += (bytes + 255) & ~(size_t)255;
        return p;
    };
    __bf16* x_bf   = (__bf16*)nxt((size_t)M_ * C_ * 2);
    __bf16* wqk_bf = (__bf16*)nxt((size_t)2048 * 2048 * 2);
    __bf16* wvg_bf = (__bf16*)nxt((size_t)4096 * 2048 * 2);
    __bf16* wo_bf  = (__bf16*)nxt((size_t)2048 * 2048 * 2);
    float*  qk_f   = (float*)nxt((size_t)M_ * 2048 * 4);
    float*  vg_f   = (float*)nxt((size_t)M_ * 4096 * 4);
    float*  qc     = (float*)nxt((size_t)BHT_ * DK_ * 4);
    float*  kc     = (float*)nxt((size_t)BHT_ * DK_ * 4);
    float*  vc     = (float*)nxt((size_t)BHT_ * DV_ * 4);
    float*  betab  = (float*)nxt((size_t)M_ * H_ * 4);
    float*  gb     = (float*)nxt((size_t)M_ * H_ * 4);
    float*  orec   = (float*)nxt((size_t)BHT_ * DV_ * 4);
    __bf16* gated  = (__bf16*)nxt((size_t)M_ * 2048 * 2);

    // 1) bf16 conversions (x and the three big weights)
    cvt_f32_bf16<<<(M_ * C_ / 4 + 255) / 256, 256, 0, stream>>>(x, x_bf, M_ * C_ / 4);
    cvt_f32_bf16<<<(2048 * 2048 / 4 + 255) / 256, 256, 0, stream>>>(qk_w, wqk_bf, 2048 * 2048 / 4);
    cvt_f32_bf16<<<(4096 * 2048 / 4 + 255) / 256, 256, 0, stream>>>(vg_w, wvg_bf, 4096 * 2048 / 4);
    cvt_f32_bf16<<<(2048 * 2048 / 4 + 255) / 256, 256, 0, stream>>>(o_w, wo_bf, 2048 * 2048 / 4);

    // 2) big GEMMs (WMMA bf16 -> f32)
    {
        dim3 g(2048 / 128, M_ / 128);
        gemm_bf16_wmma<<<g, 128, 0, stream>>>(x_bf, wqk_bf, qk_f, M_, 2048, 2048);
    }
    {
        dim3 g(4096 / 128, M_ / 128);
        gemm_bf16_wmma<<<g, 128, 0, stream>>>(x_bf, wvg_bf, vg_f, M_, 4096, 2048);
    }

    // 3) beta / g projections (fp32)
    beta_g_kernel<<<(M_ * 32) / 256, 256, 0, stream>>>(x, a_w, b_w, b_bias, beta_bias,
                                                       A_log, dt_bias, betab, gb);

    // 4) causal depthwise conv + SiLU -> (B,H,T,D) layouts
    dwconv_silu<<<(B_ * T_ * 1024) / 256, 256, 0, stream>>>(qk_f, q_conv_w, qc, 2048, 0,    1024, DK_);
    dwconv_silu<<<(B_ * T_ * 1024) / 256, 256, 0, stream>>>(qk_f, k_conv_w, kc, 2048, 1024, 1024, DK_);
    dwconv_silu<<<(B_ * T_ * 2048) / 256, 256, 0, stream>>>(vg_f, v_conv_w, vc, 4096, 0,    2048, DV_);

    // 5) l2norm(q)*SCALE, l2norm(k)
    l2norm_rows<<<BHT_ / 256, 256, 0, stream>>>(qc, BHT_, 0.125f);   // SCALE = 64^-0.5
    l2norm_rows<<<BHT_ / 256, 256, 0, stream>>>(kc, BHT_, 1.0f);

    // 6) sequential gated delta rule (register-resident state, ILP-optimized)
    delta_recurrence<<<B_ * H_, 128, 0, stream>>>(qc, kc, vc, gb, betab, orec);

    // 7) RMS norm + SiLU gate -> bf16 rows
    rms_gate<<<BHT_ / 256, 256, 0, stream>>>(orec, vg_f, gated);

    // 8) output projection (WMMA bf16 -> f32) straight into d_out
    {
        dim3 g(2048 / 128, M_ / 128);
        gemm_bf16_wmma<<<g, 128, 0, stream>>>(gated, wo_bf, out, M_, 2048, 2048);
    }
}
